// BiLSTMCRF_33337536152122
// MI455X (gfx1250) — matrix operand order repaired and checked by reading, not compiled, not run on hardware
//
#include <hip/hip_runtime.h>
#include <hip/hip_bf16.h>

// Problem constants (from reference)
#define T_SEQ 256
#define BATCH 32
#define HID   256
#define NGATE 1024      // 4*H
#define NLAB  25
#define D0    320       // E + EP
#define D1    512       // 2*H
#define ROWS  (BATCH*T_SEQ)   // 8192

typedef _Float16 v16h __attribute__((ext_vector_type(16)));
typedef _Float16 h8   __attribute__((ext_vector_type(8)));
typedef float    v8f  __attribute__((ext_vector_type(8)));

union Frag16 { v16h v; h8 h[2]; };

__device__ __forceinline__ float fast_sigmoid(float x){ return 1.0f/(1.0f+__expf(-x)); }
__device__ __forceinline__ float fast_tanh(float x){ return 1.0f - 2.0f/(__expf(2.0f*x)+1.0f); }

// ---------------- f32 -> f16 weight conversion ----------------
__global__ void f2h_kernel(const float* __restrict__ src, _Float16* __restrict__ dst, int n){
    int i = blockIdx.x*256 + threadIdx.x;
    if (i < n) dst[i] = (_Float16)src[i];
}

// ---------------- embedding gather -> f16 ----------------
__global__ void embed_kernel(const int* __restrict__ wid, const int* __restrict__ pid,
                             const float* __restrict__ Ww, const float* __restrict__ Wp,
                             _Float16* __restrict__ embH){
    int i = blockIdx.x*256 + threadIdx.x;
    if (i >= ROWS*D0) return;
    int row = i / D0, col = i - row*D0;
    float v = (col < 256) ? Ww[(long)wid[row]*256 + col]
                          : Wp[(long)pid[row]*64 + (col - 256)];
    embH[i] = (_Float16)v;
}

// ---------------- big input-projection GEMM: Z = X @ W^T (f16 in, f32 acc, f16 out) ----
// X: (8192, K) row-major f16.  W: (1024, K) row-major f16 (so B = W^T is (K,1024)).
// One 16x16 C tile per wave. grid = (512, 8, 2 dirs), block = 256 (8 waves).
__global__ __launch_bounds__(256) void gemm_xproj(
        const _Float16* __restrict__ X,
        const _Float16* __restrict__ Wf, const _Float16* __restrict__ Wb,
        _Float16* __restrict__ Zf, _Float16* __restrict__ Zb, int K){
    const _Float16* W = blockIdx.z ? Wb : Wf;
    _Float16*       Z = blockIdx.z ? Zb : Zf;
    int wave = threadIdx.x >> 5, lane = threadIdx.x & 31;
    int half_ = lane >> 4, ln = lane & 15;
    int mt = blockIdx.x;               // 0..511 row tile
    int nt = blockIdx.y*8 + wave;      // 0..63 col tile
    v8f acc = {};
    const _Float16* arow = X + (long)(mt*16 + ln)*K;
    const _Float16* brow = W + (long)(nt*16 + ln)*K;
    int kfs = K >> 5;
    for (int kf = 0; kf < kfs; ++kf){
        Frag16 a, b;
        // A 16x32 f16 layout: lane(half_,m=ln): e<8 -> K=half_*8+e ; e>=8 -> K=16+half_*8+(e-8)
        const _Float16* apt = arow + kf*32 + half_*8;
        a.h[0] = *(const h8*)apt;
        a.h[1] = *(const h8*)(apt + 16);
        // B 32x16 f16 layout: lane(half_,n=ln): e<8 -> K=half_*8+e ; e>=8 -> K=16+half_*8+(e-8)
        const _Float16* bpt = brow + kf*32 + half_*8;
        b.h[0] = *(const h8*)bpt;
        b.h[1] = *(const h8*)(bpt + 16);
        acc = __builtin_amdgcn_wmma_f32_16x16x32_f16(false, a.v, false, b.v,
                                                     (short)0, acc, false, false);
    }
    // D layout: acc[v] = D[v + 8*half_][ln]
    _Float16* zcol = Z + (long)nt*16 + ln;
    #pragma unroll
    for (int v = 0; v < 8; ++v){
        int r = mt*16 + v + 8*half_;
        zcol[(long)r*NGATE] = (_Float16)acc[v];
    }
}

// ---------------- persistent LSTM recurrence ----------------
// One workgroup per direction; 16 waves; full 256-step scan inside the kernel.
// Wave w owns gate-column tiles {w, w+16, w+32, w+48} -> each lane holds i,f,g,o
// for the same (batch,unit) and the cell update is pure register math; c stays in
// registers across all steps; h round-trips through a small padded LDS buffer.
__global__ __launch_bounds__(512) void lstm_rec_kernel(
        const _Float16* __restrict__ ZpF, const _Float16* __restrict__ ZpB,
        const _Float16* __restrict__ WhF, const _Float16* __restrict__ WhB,
        const float* __restrict__ bF, const float* __restrict__ bB,
        _Float16* __restrict__ xs /* (8192, 512) f16 */){
    int dir = blockIdx.x;
    const _Float16* Zp = dir ? ZpB : ZpF;
    const _Float16* Wh = dir ? WhB : WhF;   // (1024, 256) f16 row-major
    const float*  bias = dir ? bB  : bF;

    __shared__ alignas(16) _Float16 hbuf[BATCH][HID + 8];   // +8 halves pad: conflict-free b128 reads

    int tid = threadIdx.x;
    int wave = tid >> 5, lane = tid & 31;
    int half_ = lane >> 4, ln = lane & 15;
    int j = wave*16 + ln;                 // hidden-unit column 0..255

    for (int i = tid; i < BATCH*(HID+8); i += 512) (&hbuf[0][0])[i] = (_Float16)0.0f;
    float c_reg[16];
    #pragma unroll
    for (int q = 0; q < 16; ++q) c_reg[q] = 0.0f;
    float bias_r[4];
    #pragma unroll
    for (int g = 0; g < 4; ++g) bias_r[g] = bias[g*256 + j];
    __syncthreads();

    for (int s = 0; s < T_SEQ; ++s){
        int t = dir ? (T_SEQ - 1 - s) : s;
        v8f zero = {};
        v8f acc[2][4];
        #pragma unroll
        for (int mt = 0; mt < 2; ++mt)
            #pragma unroll
            for (int g = 0; g < 4; ++g) acc[mt][g] = zero;

        for (int kf = 0; kf < 8; ++kf){
            Frag16 a0, a1;
            const _Float16* h0 = &hbuf[ln][kf*32 + half_*8];
            a0.h[0] = *(const h8*)h0; a0.h[1] = *(const h8*)(h0 + 16);
            const _Float16* h1 = &hbuf[16 + ln][kf*32 + half_*8];
            a1.h[0] = *(const h8*)h1; a1.h[1] = *(const h8*)(h1 + 16);
            #pragma unroll
            for (int g = 0; g < 4; ++g){
                int n = g*256 + j;
                const _Float16* wr = Wh + (long)n*HID + kf*32 + half_*8;
                Frag16 b;
                b.h[0] = *(const h8*)wr; b.h[1] = *(const h8*)(wr + 16);
                acc[0][g] = __builtin_amdgcn_wmma_f32_16x16x32_f16(false, a0.v, false, b.v,
                                                                   (short)0, acc[0][g], false, false);
                acc[1][g] = __builtin_amdgcn_wmma_f32_16x16x32_f16(false, a1.v, false, b.v,
                                                                   (short)0, acc[1][g], false, false);
            }
        }
        __syncthreads();   // all waves done reading hbuf(t)

        #pragma unroll
        for (int mt = 0; mt < 2; ++mt){
            #pragma unroll
            for (int v = 0; v < 8; ++v){
                int m = mt*16 + v + 8*half_;                       // batch row
                const _Float16* zrow = Zp + ((long)(m*T_SEQ + t))*NGATE;
                float zi = acc[mt][0][v] + (float)zrow[        j] + bias_r[0];
                float zf = acc[mt][1][v] + (float)zrow[256   + j] + bias_r[1];
                float zg = acc[mt][2][v] + (float)zrow[512   + j] + bias_r[2];
                float zo = acc[mt][3][v] + (float)zrow[768   + j] + bias_r[3];
                float c = fast_sigmoid(zf)*c_reg[mt*8+v] + fast_sigmoid(zi)*fast_tanh(zg);
                c_reg[mt*8+v] = c;
                float h = fast_sigmoid(zo)*fast_tanh(c);
                hbuf[m][j] = (_Float16)h;
                xs[((long)(m*T_SEQ + t))*D1 + dir*HID + j] = (_Float16)h;
            }
        }
        __syncthreads();   // h(t+1 inputs) visible before next step's A reads
    }
}

// ---------------- output projection (512 -> 25): tiny, VALU ----------------
__global__ __launch_bounds__(256) void feats_kernel(const _Float16* __restrict__ xs1,
        const float* __restrict__ Wout, const float* __restrict__ bout,
        float* __restrict__ feats){
    int idx = blockIdx.x*256 + threadIdx.x;    // ROWS*32
    int row = idx >> 5, lab = idx & 31;
    if (row >= ROWS || lab >= NLAB) return;
    const _Float16* x = xs1 + (long)row*D1;
    float acc = bout[lab];
    #pragma unroll 4
    for (int k = 0; k < D1; ++k) acc += (float)x[k] * Wout[k*NLAB + lab];
    feats[(long)row*NLAB + lab] = acc;
}

// ---------------- Viterbi: one wave per batch ----------------
__global__ __launch_bounds__(32) void viterbi_kernel(const float* __restrict__ feats,
        const float* __restrict__ trans, unsigned char* __restrict__ bp,
        float* __restrict__ out){
    int b = blockIdx.x, j = threadIdx.x;
    __shared__ float tr[NLAB*NLAB];
    __shared__ float v[NLAB], vn[NLAB];
    for (int i = j; i < NLAB*NLAB; i += 32) tr[i] = trans[i];
    if (j < NLAB) v[j] = feats[((long)b*T_SEQ)*NLAB + j];
    __syncthreads();
    for (int t = 1; t < T_SEQ; ++t){
        if (j < NLAB){
            float best = -3.4e38f; int bi = 0;
            for (int i = 0; i < NLAB; ++i){
                float s = v[i] + tr[i*NLAB + j];
                if (s > best){ best = s; bi = i; }    // first max, like jnp.argmax
            }
            vn[j] = best + feats[((long)b*T_SEQ + t)*NLAB + j];
            bp[((long)b*T_SEQ + t)*NLAB + j] = (unsigned char)bi;
        }
        __syncthreads();
        if (j < NLAB) v[j] = vn[j];
        __syncthreads();
    }
    if (j == 0){
        float best = v[0]; int last = 0;
        for (int i = 1; i < NLAB; ++i) if (v[i] > best){ best = v[i]; last = i; }
        out[b] = best;
        // reference path shape is (B, T+1): path[0]=0 (zero bp row), path[k]=tag_{k-1}
        float* path = out + BATCH + (long)b*(T_SEQ + 1);
        path[T_SEQ] = (float)last;
        int cur = last;
        for (int t = T_SEQ - 1; t >= 1; --t){
            cur = bp[((long)b*T_SEQ + t)*NLAB + cur];
            path[t] = (float)cur;
        }
        path[0] = 0.0f;
    }
}

extern "C" void kernel_launch(void* const* d_in, const int* in_sizes, int n_in,
                              void* d_out, int out_size, void* d_ws, size_t ws_size,
                              hipStream_t stream) {
    (void)n_in; (void)out_size; (void)ws_size;

    // ---- resolve flattened-input ordering (insertion / sorted-dict / fully-sorted pytree) ----
    int I_word, I_pos, I_Wword, I_Wpos, I_Wout, I_bout, I_trans;
    int I_Wih[2][2], I_Whh[2][2], I_b[2][2];            // [layer][0=f,1=b]
    if (in_sizes[0] == 12800) {
        // fully sorted pytree: W_out, W_pos, W_word, b_out, lengths, lstm[...], pos_ids, transitions, word_ids
        I_Wout = 0; I_Wpos = 1; I_Wword = 2; I_bout = 3; /* lengths = 4 */
        I_Whh[0][1] = 5;  I_Whh[0][0] = 6;  I_Wih[0][1] = 7;  I_Wih[0][0] = 8;  I_b[0][1] = 9;  I_b[0][0] = 10;
        I_Whh[1][1] = 11; I_Whh[1][0] = 12; I_Wih[1][1] = 13; I_Wih[1][0] = 14; I_b[1][1] = 15; I_b[1][0] = 16;
        I_pos = 17; I_trans = 18; I_word = 19;
    } else {
        I_word = 0; I_pos = 1; /* lengths = 2 */ I_Wword = 3; I_Wpos = 4;
        if (in_sizes[5] == 327680) {   // insertion-ordered dicts
            I_Wih[0][0] = 5;  I_Whh[0][0] = 6;  I_b[0][0] = 7;  I_Wih[0][1] = 8;  I_Whh[0][1] = 9;  I_b[0][1] = 10;
            I_Wih[1][0] = 11; I_Whh[1][0] = 12; I_b[1][0] = 13; I_Wih[1][1] = 14; I_Whh[1][1] = 15; I_b[1][1] = 16;
        } else {                       // alphabetically sorted dict keys
            I_Whh[0][1] = 5;  I_Whh[0][0] = 6;  I_Wih[0][1] = 7;  I_Wih[0][0] = 8;  I_b[0][1] = 9;  I_b[0][0] = 10;
            I_Whh[1][1] = 11; I_Whh[1][0] = 12; I_Wih[1][1] = 13; I_Wih[1][0] = 14; I_b[1][1] = 15; I_b[1][0] = 16;
        }
        I_Wout = 17; I_bout = 18; I_trans = 19;
    }

    // ---- workspace layout ----
    char* ws = (char*)d_ws;
    size_t off = 0;
    auto take = [&](size_t bytes) -> void* {
        void* p = ws + off;
        off = (off + bytes + 255) & ~(size_t)255;
        return p;
    };
    _Float16* embH = (_Float16*)take((size_t)ROWS*D0*2);
    _Float16* xs0  = (_Float16*)take((size_t)ROWS*D1*2);
    _Float16* xs1  = (_Float16*)take((size_t)ROWS*D1*2);
    _Float16* ZpF  = (_Float16*)take((size_t)ROWS*NGATE*2);
    _Float16* ZpB  = (_Float16*)take((size_t)ROWS*NGATE*2);
    _Float16* Wih0h[2], *Whh0h[2], *Wih1h[2], *Whh1h[2];
    for (int d = 0; d < 2; ++d) Wih0h[d] = (_Float16*)take((size_t)NGATE*D0*2);
    for (int d = 0; d < 2; ++d) Whh0h[d] = (_Float16*)take((size_t)NGATE*HID*2);
    for (int d = 0; d < 2; ++d) Wih1h[d] = (_Float16*)take((size_t)NGATE*D1*2);
    for (int d = 0; d < 2; ++d) Whh1h[d] = (_Float16*)take((size_t)NGATE*HID*2);
    float* featsBuf = (float*)take((size_t)ROWS*NLAB*4);
    unsigned char* bpBuf = (unsigned char*)take((size_t)ROWS*NLAB);

    auto conv = [&](int idx, _Float16* dst, int n){
        f2h_kernel<<<(n + 255)/256, 256, 0, stream>>>((const float*)d_in[idx], dst, n);
    };
    conv(I_Wih[0][0], Wih0h[0], NGATE*D0);  conv(I_Wih[0][1], Wih0h[1], NGATE*D0);
    conv(I_Whh[0][0], Whh0h[0], NGATE*HID); conv(I_Whh[0][1], Whh0h[1], NGATE*HID);
    conv(I_Wih[1][0], Wih1h[0], NGATE*D1);  conv(I_Wih[1][1], Wih1h[1], NGATE*D1);
    conv(I_Whh[1][0], Whh1h[0], NGATE*HID); conv(I_Whh[1][1], Whh1h[1], NGATE*HID);

    embed_kernel<<<(ROWS*D0 + 255)/256, 256, 0, stream>>>(
        (const int*)d_in[I_word], (const int*)d_in[I_pos],
        (const float*)d_in[I_Wword], (const float*)d_in[I_Wpos], embH);

    // Layer 0
    gemm_xproj<<<dim3(512, 8, 2), 256, 0, stream>>>(embH, Wih0h[0], Wih0h[1], ZpF, ZpB, D0);
    lstm_rec_kernel<<<2, 512, 0, stream>>>(ZpF, ZpB, Whh0h[0], Whh0h[1],
        (const float*)d_in[I_b[0][0]], (const float*)d_in[I_b[0][1]], xs0);
    // Layer 1
    gemm_xproj<<<dim3(512, 8, 2), 256, 0, stream>>>(xs0, Wih1h[0], Wih1h[1], ZpF, ZpB, D1);
    lstm_rec_kernel<<<2, 512, 0, stream>>>(ZpF, ZpB, Whh1h[0], Whh1h[1],
        (const float*)d_in[I_b[1][0]], (const float*)d_in[I_b[1][1]], xs1);

    feats_kernel<<<(ROWS*32)/256, 256, 0, stream>>>(xs1,
        (const float*)d_in[I_Wout], (const float*)d_in[I_bout], featsBuf);

    viterbi_kernel<<<BATCH, 32, 0, stream>>>(featsBuf,
        (const float*)d_in[I_trans], bpBuf, (float*)d_out);
}
